// OptimizedStatePropagator_584115553022
// MI455X (gfx1250) — compile-verified, matrix operand order checked by reading, not run on hardware
//
#include <hip/hip_runtime.h>
#include <hip/hip_bf16.h>

// MI455X / gfx1250, wave32. bf16 WMMA (16x16x32) + fp32 accumulation for all
// GEMMs; fp32 softmax/LayerNorm/sigmoid/tanh. Big GEMMs use a block-cooperative
// 128x256 LDS tile filled with GLOBAL_LOAD_ASYNC_TO_LDS_B128 (double-buffered,
// s_wait_asynccnt + barrier), skinny attention GEMMs use per-wave 64x64 tiles.

typedef __attribute__((ext_vector_type(16))) __bf16 v16bf;
typedef __attribute__((ext_vector_type(8)))  __bf16 v8bf;
typedef __attribute__((ext_vector_type(8)))  float  v8f;

#if defined(__AMDGCN__) && defined(__gfx1250__)
#define USE_ASYNC_LDS 1
#else
#define USE_ASYNC_LDS 0
#endif

// ---------------------------------------------------------------------------
// 16-byte global->LDS copy. Async path: per-lane DMA, tracked by ASYNCcnt, no
// VGPR round trip. Generic LDS pointer's low 32 bits are the LDS address
// (flat aperture truncation rule, ISA 10.2).
// ---------------------------------------------------------------------------
__device__ __forceinline__ void copy16_g2l(__bf16* lds, const __bf16* g) {
#if USE_ASYNC_LDS
  unsigned lds_off = (unsigned)(unsigned long long)lds;
  asm volatile("global_load_async_to_lds_b128 %0, %1, off"
               :: "v"(lds_off), "v"(g) : "memory");
#else
  *(uint4*)lds = *(const uint4*)g;
#endif
}
__device__ __forceinline__ void wait_async_le6() {   // prev stage (6 ops) done
#if USE_ASYNC_LDS
  asm volatile("s_wait_asynccnt 0x6" ::: "memory");
#endif
}
__device__ __forceinline__ void wait_async_0() {
#if USE_ASYNC_LDS
  asm volatile("s_wait_asynccnt 0x0" ::: "memory");
#endif
}

// ---------------------------------------------------------------------------
// Per-lane A/B fragment loader for V_WMMA_F32_16X16X32_BF16 (ISA 7.12.2):
// lanes 0-15 rows 0-15 K-base 0; lanes 16-31 rows 0-15 K-base 8;
// VGPR0-3 <- K kb+0..7, VGPR4-7 <- K kb+16..23. Two 16B vector loads.
// Works for global (ld=K) and LDS (ld=LDST) K-contiguous operands.
// ---------------------------------------------------------------------------
__device__ __forceinline__ v16bf load_frag(const __bf16* __restrict__ p16) {
  v8bf lo = *(const v8bf*)(p16);
  v8bf hi = *(const v8bf*)(p16 + 16);
  v16bf f;
#pragma unroll
  for (int i = 0; i < 8; ++i) { f[i] = lo[i]; f[i + 8] = hi[i]; }
  return f;
}

// shared epilogue: lane 0-15 -> N=lane, M=e ; lane 16-31 -> N=lane-16, M=8+e
__device__ __forceinline__ void store_tile(
    v8f acc[4][4], int tm, int tn, int lane, float alpha,
    const float* __restrict__ bias, float* __restrict__ Cf,
    __bf16* __restrict__ Cbf, __bf16* __restrict__ CbfT, long sTb, int N)
{
  const int col = lane & 15;
  const int rb  = (lane & 16) ? 8 : 0;
#pragma unroll
  for (int i = 0; i < 4; ++i)
#pragma unroll
    for (int j = 0; j < 4; ++j)
#pragma unroll
      for (int e = 0; e < 8; ++e) {
        int m = tm + 16 * i + rb + e;
        int n = tn + 16 * j + col;
        float v = acc[i][j][e] * alpha;
        if (bias) v += bias[n];
        long idx = (long)m * N + n;
        if (Cf)   Cf[idx]  = v;
        if (Cbf)  Cbf[idx] = (__bf16)v;
        if (CbfT) // per-batch transpose: batch = m>>11 (S=2048)
          CbfT[((long)(m >> 11)) * sTb + (long)n * 2048 + (m & 2047)] = (__bf16)v;
      }
}

// ---------------------------------------------------------------------------
// Block-cooperative GEMM: C[MxN] = alpha*A[MxK]*Bt[NxK]^T (+bias). 256 threads
// = 8 waves compute a 128x256 tile (2x4 waves of 64x64). K staged in 32-wide
// double-buffered LDS tiles via async copies. Row stride padded to 40 bf16
// (80 B) so the 16 fragment lanes' ds_load_b128 tile all 64 banks.
// Requires M%128==0, N%256==0, K%32==0.
// ---------------------------------------------------------------------------
#define LDST 40

__global__ __launch_bounds__(256) void wmma_gemm_block(
    const __bf16* __restrict__ A, const __bf16* __restrict__ Bt,
    float* __restrict__ Cf, __bf16* __restrict__ Cbf,
    __bf16* __restrict__ CbfT, long sTb,
    const float* __restrict__ bias, float alpha, int M, int N, int K)
{
  __shared__ __align__(16) __bf16 smA[2][128 * LDST];
  __shared__ __align__(16) __bf16 smB[2][256 * LDST];

  const int nBn = N >> 8;
  const int tmB = ((int)blockIdx.x / nBn) << 7;
  const int tnB = ((int)blockIdx.x % nBn) << 8;

  const int t    = threadIdx.x;
  const int warp = t >> 5;
  const int lane = t & 31;
  const int r    = lane & 15;
  const int kb   = (lane & 16) ? 8 : 0;
  const int wm   = (warp >> 2) << 6;   // 0 / 64
  const int wn   = (warp & 3) << 6;    // 0..192

  v8f acc[4][4];
#pragma unroll
  for (int i = 0; i < 4; ++i)
#pragma unroll
    for (int j = 0; j < 4; ++j)
#pragma unroll
      for (int e = 0; e < 8; ++e) acc[i][j][e] = 0.0f;

  // stage one 32-wide K slice: A 128x32 (512 16B chunks), B 256x32 (1024)
  auto stage = [&](int buf, int k0) {
#pragma unroll
    for (int q = 0; q < 2; ++q) {
      int c = t + 256 * q, row = c >> 2, cc = c & 3;
      copy16_g2l(&smA[buf][row * LDST + cc * 8],
                 A + (long)(tmB + row) * K + k0 + cc * 8);
    }
#pragma unroll
    for (int q = 0; q < 4; ++q) {
      int c = t + 256 * q, row = c >> 2, cc = c & 3;
      copy16_g2l(&smB[buf][row * LDST + cc * 8],
                 Bt + (long)(tnB + row) * K + k0 + cc * 8);
    }
  };

  const int nsteps = K >> 5;
  int buf = 0;
  stage(0, 0);
  for (int step = 0; step < nsteps; ++step) {
    const bool have = (step + 1) < nsteps;
    if (have) stage(buf ^ 1, (step + 1) << 5);
    if (have) wait_async_le6(); else wait_async_0();   // current buffer landed
    __syncthreads();

    const __bf16* sA = smA[buf];
    const __bf16* sB = smB[buf];
    v16bf af[4], bq[4];
#pragma unroll
    for (int i = 0; i < 4; ++i)
      af[i] = load_frag(sA + (wm + 16 * i + r) * LDST + kb);
#pragma unroll
    for (int j = 0; j < 4; ++j)
      bq[j] = load_frag(sB + (wn + 16 * j + r) * LDST + kb);
#pragma unroll
    for (int i = 0; i < 4; ++i)
#pragma unroll
      for (int j = 0; j < 4; ++j)
        acc[i][j] = __builtin_amdgcn_wmma_f32_16x16x32_bf16(
            false, af[i], false, bq[j], (short)0, acc[i][j], false, false);

    __syncthreads();  // all waves done reading buf before it is restaged
    buf ^= 1;
  }

  store_tile(acc, tmB + wm, tnB + wn, lane, alpha, bias, Cf, Cbf, CbfT, sTb, N);
}

// ---------------------------------------------------------------------------
// Per-wave 64x64 GEMM (for skinny M=64 attention GEMMs), batched via grid.y.
// ---------------------------------------------------------------------------
__global__ __launch_bounds__(256) void wmma_gemm_wave(
    const __bf16* __restrict__ A, long sAb,
    const __bf16* __restrict__ Bt, long sBb,
    float* __restrict__ Cf, long sCb,
    float alpha, int M, int N, int K)
{
  const int tilesN = N >> 6;
  const int tilesM = M >> 6;
  const int wave = blockIdx.x * (blockDim.x >> 5) + (threadIdx.x >> 5);
  if (wave >= tilesM * tilesN) return;       // wave-uniform: EXEC stays all-1s
  const int tm = (wave / tilesN) << 6;
  const int tn = (wave % tilesN) << 6;
  const int lane = threadIdx.x & 31;
  const int r  = lane & 15;
  const int kb = (lane & 16) ? 8 : 0;

  const __bf16* Ab = A  + (long)blockIdx.y * sAb;
  const __bf16* Bb = Bt + (long)blockIdx.y * sBb;

  v8f acc[4][4];
#pragma unroll
  for (int i = 0; i < 4; ++i)
#pragma unroll
    for (int j = 0; j < 4; ++j)
#pragma unroll
      for (int e = 0; e < 8; ++e) acc[i][j][e] = 0.0f;

  const __bf16* aBase = Ab + (long)(tm + r) * K + kb;
  const __bf16* bBase = Bb + (long)(tn + r) * K + kb;

  for (int k0 = 0; k0 < K; k0 += 32) {
    if (k0 + 32 < K) {
      __builtin_prefetch((const void*)(aBase + k0 + 32), 0, 3);
      __builtin_prefetch((const void*)(bBase + k0 + 32), 0, 3);
    }
    v16bf af[4], bq[4];
#pragma unroll
    for (int i = 0; i < 4; ++i) af[i] = load_frag(aBase + (long)(16 * i) * K + k0);
#pragma unroll
    for (int j = 0; j < 4; ++j) bq[j] = load_frag(bBase + (long)(16 * j) * K + k0);
#pragma unroll
    for (int i = 0; i < 4; ++i)
#pragma unroll
      for (int j = 0; j < 4; ++j)
        acc[i][j] = __builtin_amdgcn_wmma_f32_16x16x32_bf16(
            false, af[i], false, bq[j], (short)0, acc[i][j], false, false);
  }
  store_tile(acc, tm, tn, lane, alpha, nullptr,
             Cf ? Cf + (long)blockIdx.y * sCb : nullptr, nullptr, nullptr, 0, N);
}

// ---------------------------------------------------------------------------
// Helpers: fp32 -> bf16 convert / transpose-convert, block reductions.
// ---------------------------------------------------------------------------
__global__ void cvt_f32_bf16(const float* __restrict__ x, __bf16* __restrict__ y, long n) {
  long stride = (long)gridDim.x * blockDim.x;
  for (long i = (long)blockIdx.x * blockDim.x + threadIdx.x; i < n; i += stride)
    y[i] = (__bf16)x[i];
}

__global__ void transpose_cvt(const float* __restrict__ W, __bf16* __restrict__ Wt,
                              int K, int N) {
  long total = (long)K * N;
  long stride = (long)gridDim.x * blockDim.x;
  for (long i = (long)blockIdx.x * blockDim.x + threadIdx.x; i < total; i += stride) {
    int k = (int)(i / N), n = (int)(i % N);
    Wt[(long)n * K + k] = (__bf16)W[i];
  }
}

__device__ __forceinline__ float blk_sum(float v, float* red) {
  int t = threadIdx.x;
  red[t] = v; __syncthreads();
  for (int off = 128; off > 0; off >>= 1) { if (t < off) red[t] += red[t + off]; __syncthreads(); }
  float r = red[0]; __syncthreads();
  return r;
}
__device__ __forceinline__ float blk_max(float v, float* red) {
  int t = threadIdx.x;
  red[t] = v; __syncthreads();
  for (int off = 128; off > 0; off >>= 1) { if (t < off) red[t] = fmaxf(red[t], red[t + off]); __syncthreads(); }
  float r = red[0]; __syncthreads();
  return r;
}

// softmax over S=2048 per (b,n) row; writes bf16 attn
__global__ __launch_bounds__(256) void softmax_rows(const float* __restrict__ S,
                                                    __bf16* __restrict__ A) {
  __shared__ float red[256];
  const int row = blockIdx.x, t = threadIdx.x;
  const float* s = S + (long)row * 2048;
  float v[8]; float mx = -3.4e38f;
#pragma unroll
  for (int k = 0; k < 8; ++k) { v[k] = s[t + 256 * k]; mx = fmaxf(mx, v[k]); }
  mx = blk_max(mx, red);
  float sum = 0.0f;
#pragma unroll
  for (int k = 0; k < 8; ++k) { v[k] = __expf(v[k] - mx); sum += v[k]; }
  sum = blk_sum(sum, red);
  float inv = 1.0f / sum;
  __bf16* a = A + (long)row * 2048;
#pragma unroll
  for (int k = 0; k < 8; ++k) a[t + 256 * k] = (__bf16)(v[k] * inv);
}

// LN(prev_states) and LN(routed) -> bf16 concat [ps_n | in_n]; second half of
// reset_in pre-filled with in_n; ps_n kept fp32.
__global__ __launch_bounds__(256) void ln_concat(
    const float* __restrict__ prev, const float* __restrict__ routed,
    const float* __restrict__ gs, const float* __restrict__ bs,
    const float* __restrict__ gi, const float* __restrict__ bi,
    float* __restrict__ psn, __bf16* __restrict__ concat,
    __bf16* __restrict__ resetin)
{
  __shared__ float red[256];
  const int row = blockIdx.x, t = threadIdx.x;
  const float invD = 1.0f / 1024.0f;

  const float* x = prev + (long)row * 1024;
  float v[4], s1 = 0.f, s2 = 0.f;
#pragma unroll
  for (int k = 0; k < 4; ++k) { v[k] = x[t + 256 * k]; s1 += v[k]; s2 += v[k] * v[k]; }
  s1 = blk_sum(s1, red); s2 = blk_sum(s2, red);
  float m = s1 * invD, rstd = rsqrtf(s2 * invD - m * m + 1e-5f);
#pragma unroll
  for (int k = 0; k < 4; ++k) {
    int d = t + 256 * k;
    float y = (v[k] - m) * rstd * gs[d] + bs[d];
    psn[(long)row * 1024 + d] = y;
    concat[(long)row * 2048 + d] = (__bf16)y;
  }

  x = routed + (long)row * 1024;
  s1 = 0.f; s2 = 0.f;
#pragma unroll
  for (int k = 0; k < 4; ++k) { v[k] = x[t + 256 * k]; s1 += v[k]; s2 += v[k] * v[k]; }
  s1 = blk_sum(s1, red); s2 = blk_sum(s2, red);
  m = s1 * invD; rstd = rsqrtf(s2 * invD - m * m + 1e-5f);
#pragma unroll
  for (int k = 0; k < 4; ++k) {
    int d = t + 256 * k;
    float y = (v[k] - m) * rstd * gi[d] + bi[d];
    concat [(long)row * 2048 + 1024 + d] = (__bf16)y;
    resetin[(long)row * 2048 + 1024 + d] = (__bf16)y;
  }
}

// u = sigmoid(Upre), r = sigmoid(Rpre); reset_in[:,0:D] = bf16(r * ps_n)
__global__ void gates_elem(const float* __restrict__ Upre, const float* __restrict__ Rpre,
                           const float* __restrict__ psn,
                           float* __restrict__ u, __bf16* __restrict__ resetin) {
  long stride = (long)gridDim.x * blockDim.x;
  for (long i = (long)blockIdx.x * blockDim.x + threadIdx.x; i < 1048576L; i += stride) {
    long m = i >> 10; int d = (int)(i & 1023);
    float uu = 1.0f / (1.0f + __expf(-Upre[i]));
    float rr = 1.0f / (1.0f + __expf(-Rpre[i]));
    u[i] = uu;
    resetin[m * 2048 + d] = (__bf16)(rr * psn[i]);
  }
}

// cand = tanh(Npre); new = (1-u)*prev + u*cand; out = LN(new)*g_out + b_out
__global__ __launch_bounds__(256) void final_ln(
    const float* __restrict__ Npre, const float* __restrict__ u,
    const float* __restrict__ prev, const float* __restrict__ go,
    const float* __restrict__ bo, float* __restrict__ out)
{
  __shared__ float red[256];
  const int row = blockIdx.x, t = threadIdx.x;
  const float invD = 1.0f / 1024.0f;
  float v[4], s1 = 0.f, s2 = 0.f;
#pragma unroll
  for (int k = 0; k < 4; ++k) {
    long i = (long)row * 1024 + t + 256 * k;
    float cand = tanhf(Npre[i]);
    float uu = u[i];
    float nv = (1.0f - uu) * prev[i] + uu * cand;
    v[k] = nv; s1 += nv; s2 += nv * nv;
  }
  s1 = blk_sum(s1, red); s2 = blk_sum(s2, red);
  float m = s1 * invD, rstd = rsqrtf(s2 * invD - m * m + 1e-5f);
#pragma unroll
  for (int k = 0; k < 4; ++k) {
    int d = t + 256 * k;
    long i = (long)row * 1024 + d;
    out[i] = (v[k] - m) * rstd * go[d] + bo[d];
  }
}

// ---------------------------------------------------------------------------
extern "C" void kernel_launch(void* const* d_in, const int* in_sizes, int n_in,
                              void* d_out, int out_size, void* d_ws, size_t ws_size,
                              hipStream_t stream) {
  (void)in_sizes; (void)n_in; (void)out_size; (void)ws_size;
  const float* prev      = (const float*)d_in[0];   // (16,64,1024)
  const float* tokens    = (const float*)d_in[1];   // (16,2048,768)
  const float* W_proj    = (const float*)d_in[2];   // (768,1024)
  const float* b_proj    = (const float*)d_in[3];
  const float* g_state   = (const float*)d_in[4];
  const float* beta_state= (const float*)d_in[5];
  const float* g_input   = (const float*)d_in[6];
  const float* beta_input= (const float*)d_in[7];
  const float* g_out     = (const float*)d_in[8];
  const float* beta_out  = (const float*)d_in[9];
  const float* W_u = (const float*)d_in[10]; const float* b_u = (const float*)d_in[11];
  const float* W_r = (const float*)d_in[12]; const float* b_r = (const float*)d_in[13];
  const float* W_n = (const float*)d_in[14]; const float* b_n = (const float*)d_in[15];

  char* ws = (char*)d_ws; size_t o = 0;
  auto take = [&](size_t bytes) -> void* {
    void* p = ws + o; o += (bytes + 255) & ~(size_t)255; return p;
  };
  __bf16* tokA   = (__bf16*)take(25165824L * 2);  // tokens bf16 (32768 x 768)
  __bf16* psA    = (__bf16*)take( 1048576L * 2);  // prev_states bf16
  __bf16* WpT    = (__bf16*)take(  786432L * 2);  // W_proj^T bf16 (1024 x 768)
  __bf16* WuT    = (__bf16*)take( 2097152L * 2);  // W_u^T bf16 (1024 x 2048)
  __bf16* WrT    = (__bf16*)take( 2097152L * 2);
  __bf16* WnT    = (__bf16*)take( 2097152L * 2);
  __bf16* tokB   = (__bf16*)take(33554432L * 2);  // tok bf16 (32768 x 1024)
  __bf16* tokT   = (__bf16*)take(33554432L * 2);  // tok^T bf16 per batch (1024 x 2048)
  float*  scores = (float*) take( 2097152L * 4);  // (16,64,2048)
  __bf16* attn   = (__bf16*)take( 2097152L * 2);
  float*  routed = (float*) take( 1048576L * 4);
  float*  psn    = (float*) take( 1048576L * 4);
  __bf16* concat = (__bf16*)take( 2097152L * 2);  // (1024 x 2048)
  __bf16* resetin= (__bf16*)take( 2097152L * 2);
  float*  Upre   = (float*) take( 1048576L * 4);
  float*  Rpre   = (float*) take( 1048576L * 4);
  float*  Npre   = (float*) take( 1048576L * 4);
  float*  ubuf   = (float*) take( 1048576L * 4);

  // ---- operand down-conversion / weight transposes (K-contiguous bf16) ----
  cvt_f32_bf16 <<<2048, 256, 0, stream>>>(tokens, tokA, 25165824L);
  cvt_f32_bf16 <<< 512, 256, 0, stream>>>(prev,   psA,   1048576L);
  transpose_cvt<<<1024, 256, 0, stream>>>(W_proj, WpT,  768, 1024);
  transpose_cvt<<<2048, 256, 0, stream>>>(W_u,    WuT, 2048, 1024);
  transpose_cvt<<<2048, 256, 0, stream>>>(W_r,    WrT, 2048, 1024);
  transpose_cvt<<<2048, 256, 0, stream>>>(W_n,    WnT, 2048, 1024);

  // ---- tok = tokens @ W_proj + b_proj : (32768x768)x(768x1024), dominant
  //      GEMM; epilogue emits bf16 tok and per-batch bf16 tok^T
  wmma_gemm_block<<<1024, 256, 0, stream>>>(
      tokA, WpT, nullptr, tokB, tokT, 2048L * 1024, b_proj,
      1.0f, 32768, 1024, 768);

  // ---- scores = ps @ tok^T / sqrt(D) : per batch (64x1024)x(1024x2048)
  wmma_gemm_wave<<<dim3(4, 16), 256, 0, stream>>>(
      psA, 64L * 1024, tokB, 2048L * 1024, scores, 64L * 2048,
      0.03125f, 64, 2048, 1024);

  softmax_rows<<<1024, 256, 0, stream>>>(scores, attn);

  // ---- routed = attn @ tok : per batch (64x2048)x(2048x1024)
  wmma_gemm_wave<<<dim3(2, 16), 256, 0, stream>>>(
      attn, 64L * 2048, tokT, 1024L * 2048, routed, 64L * 1024,
      1.0f, 64, 1024, 2048);

  ln_concat<<<1024, 256, 0, stream>>>(prev, routed, g_state, beta_state,
                                      g_input, beta_input, psn, concat, resetin);

  // ---- GRU gate GEMMs: (1024x2048)x(2048x1024), bias folded in epilogue ----
  wmma_gemm_block<<<32, 256, 0, stream>>>(concat, WuT, Upre, nullptr,
      nullptr, 0, b_u, 1.0f, 1024, 1024, 2048);
  wmma_gemm_block<<<32, 256, 0, stream>>>(concat, WrT, Rpre, nullptr,
      nullptr, 0, b_r, 1.0f, 1024, 1024, 2048);

  gates_elem<<<1024, 256, 0, stream>>>(Upre, Rpre, psn, ubuf, resetin);

  wmma_gemm_block<<<32, 256, 0, stream>>>(resetin, WnT, Npre, nullptr,
      nullptr, 0, b_n, 1.0f, 1024, 1024, 2048);

  final_ln<<<1024, 256, 0, stream>>>(Npre, ubuf, prev, g_out, beta_out,
                                     (float*)d_out);
}
